// PNALayer_47725676593250
// MI455X (gfx1250) — matrix operand order repaired and checked by reading, not compiled, not run on hardware
//
#include <hip/hip_runtime.h>
#include <math.h>

// ---------------- problem constants ----------------
#define N_NODES 50000
#define N_EDGES 800000
#define DIM     96
#define NTOW    4
#define TI      24
#define KIN_M   48      // edge GEMM K (2*TI)
#define KIN_U   312     // node GEMM K ((4*3+1)*TI)
#define DELTA   2.833f
#define BN_EPS  1e-5f
#define NEG     0.01f

typedef __attribute__((ext_vector_type(16))) _Float16 v16h;
typedef __attribute__((ext_vector_type(8)))  float    v8f;

union F16x16 { v16h v; _Float16 h[16]; unsigned u[8]; };
union H4     { _Float16 h[4]; uint2 u2; };

// D = A(16x32 f16) * B(32x16 f16) + C(16x16 f32)
static __device__ inline v8f wmma16(v16h a, v16h b, v8f c) {
  return __builtin_amdgcn_wmma_f32_16x16x32_f16(
      /*neg_a=*/false, a, /*neg_b=*/false, b,
      /*c_mod=*/(short)0, c, /*reuse_a=*/false, /*reuse_b=*/false);
}

// monotonic uint key for float (deterministic atomic max/min on floats)
static __device__ inline unsigned f2k(float f) {
  unsigned u = __float_as_uint(f);
  return (u >> 31) ? ~u : (u | 0x80000000u);
}
static __device__ inline float k2f(unsigned k) {
  unsigned u = (k >> 31) ? (k ^ 0x80000000u) : ~k;
  return __uint_as_float(u);
}
static __device__ inline float leaky(float v) { return v > 0.f ? v : NEG * v; }

// 16-bit A layout (contiguous, matches B): vgpr v holds K pair base:
//   K0 = 2*v + (laneHi<<4)  (+32*chunk)
static __device__ inline int a_k0(int v, int hi, int c) {
  return 2 * v + (hi << 4) + (c << 5);
}

// B fragment (32x16) from GLOBAL f32 with strides + zero padding:
//   vgpr v, lane l: K = 2v + 16*(l>>4) (+32*chunk), N = nt*16 + (l&15)
static __device__ inline v16h load_bfrag(const float* __restrict__ W, int c, int nt,
                                         int Kmax, int Nmax, int sK, int sN, int lane) {
  F16x16 f;
  const int hi = lane >> 4;
  const int n  = nt * 16 + (lane & 15);
#pragma unroll
  for (int v = 0; v < 8; ++v) {
    int K = 2 * v + hi * 16 + c * 32;
    float x0 = (K     < Kmax && n < Nmax) ? W[(size_t)K * sK + (size_t)n * sN] : 0.f;
    float x1 = (K + 1 < Kmax && n < Nmax) ? W[(size_t)(K + 1) * sK + (size_t)n * sN] : 0.f;
    f.h[2 * v]     = (_Float16)x0;
    f.h[2 * v + 1] = (_Float16)x1;
  }
  return f.v;
}

// B fragment from LDS f16 staged K-major: Bs[n * rowStride + K] (K pairs contiguous)
static __device__ inline v16h ldb_lds(const _Float16* Bs, int c, int nt, int rowStride, int lane) {
  F16x16 f;
  const int hi = lane >> 4;
  const int n  = nt * 16 + (lane & 15);
#pragma unroll
  for (int v = 0; v < 8; ++v) {
    int K = 2 * v + hi * 16 + c * 32;
    f.u[v] = *(const unsigned*)(Bs + n * rowStride + K);
  }
  return f.v;
}

// ---------------- kernel 1: init workspace ----------------
__global__ __launch_bounds__(256) void init_kernel(float* __restrict__ sum, float* __restrict__ sq,
                                                   unsigned* __restrict__ mx, unsigned* __restrict__ mn,
                                                   float* __restrict__ deg) {
  const size_t n = (size_t)N_NODES * DIM;
  for (size_t i = blockIdx.x * (size_t)blockDim.x + threadIdx.x; i < n;
       i += (size_t)gridDim.x * blockDim.x) {
    sum[i] = 0.f; sq[i] = 0.f;
    mx[i] = 0u;            // minimum unsigned key
    mn[i] = 0xFFFFFFFFu;   // maximum unsigned key
    if (i < N_NODES) deg[i] = 0.f;
  }
}

// ---------------- kernel 2: degree ----------------
__global__ __launch_bounds__(256) void deg_kernel(const int* __restrict__ dst, float* __restrict__ deg) {
  for (size_t e = blockIdx.x * (size_t)blockDim.x + threadIdx.x; e < N_EDGES;
       e += (size_t)gridDim.x * blockDim.x)
    atomicAdd(&deg[dst[e]], 1.f);
}

// ---------------- kernel 3: fused edge message GEMM + segment aggregation ----------------
#define EWAVES 4
#define EROW   200   // f16 feature row stride (400B -> conflict-free bank stride)
#define BROW   66    // f16 weight row stride (132B -> odd bank stride, conflict-free)
__global__ __launch_bounds__(EWAVES * 32) void edge_kernel(
    const float* __restrict__ h, const int* __restrict__ src, const int* __restrict__ dst,
    const float* __restrict__ M_w, const float* __restrict__ M_b,
    float* __restrict__ sum, float* __restrict__ sq,
    unsigned* __restrict__ mx, unsigned* __restrict__ mn) {
  __shared__ __align__(16) _Float16 featS[EWAVES][16 * EROW];   // 25.6 KB
  __shared__ __align__(16) _Float16 BwS[NTOW][32 * BROW];       // 16.9 KB

  const int lane = threadIdx.x & 31;
  const int wave = threadIdx.x >> 5;
  const int hi = lane >> 4, lo = lane & 15;

  // stage M_w once per block into LDS as f16, layout [t][n(32)][K(64)], zero padded
  for (int idx = threadIdx.x; idx < NTOW * 32 * 64; idx += EWAVES * 32) {
    int K = idx & 63;
    int n = (idx >> 6) & 31;
    int t = idx >> 11;
    float v = (K < KIN_M && n < TI) ? M_w[((size_t)t * KIN_M + K) * TI + n] : 0.f;
    BwS[t][n * BROW + K] = (_Float16)v;
  }
  __syncthreads();

  float bias0[NTOW], bias1[NTOW];
#pragma unroll
  for (int t = 0; t < NTOW; ++t) {
    bias0[t] = M_b[t * TI + lo];
    bias1[t] = (lo < 8) ? M_b[t * TI + 16 + lo] : 0.f;
  }

  const float4* __restrict__ h4 = (const float4*)h;
  const int ntiles = N_EDGES / 16;
  for (int tile = blockIdx.x * EWAVES + wave; tile < ntiles; tile += gridDim.x * EWAVES) {
    const int base = tile * 16;
    const int sReg = src[base + lo];
    const int dReg = dst[base + lo];

    // stage src|dst rows (f32->f16) into LDS: row m = [src 96 | dst 96]
#pragma unroll
    for (int i = 0; i < 24; ++i) {
      int c = lane + 32 * i;          // 768 float4 chunks
      int m = c / 48, p = c % 48;
      int row  = (p < 24) ? __shfl(sReg, m, 32) : __shfl(dReg, m, 32);
      int colf = (p < 24) ? p : p - 24;
      float4 v = h4[(size_t)row * 24 + colf];
      H4 cv;
      cv.h[0] = (_Float16)v.x; cv.h[1] = (_Float16)v.y;
      cv.h[2] = (_Float16)v.z; cv.h[3] = (_Float16)v.w;
      int off = m * EROW + ((p < 24) ? 0 : 96) + colf * 4;
      *(uint2*)&featS[wave][off] = cv.u2;
    }
    // wave-internal LDS RAW is in-order (DScnt); no barrier (per-wave region)

    const _Float16* rowp = &featS[wave][lo * EROW];
#pragma unroll
    for (int t = 0; t < NTOW; ++t) {
      v8f acc0, acc1;
#pragma unroll
      for (int r = 0; r < 8; ++r) { acc0[r] = bias0[t]; acc1[r] = bias1[t]; }

      F16x16 A0, A1;
#pragma unroll
      for (int v = 0; v < 8; ++v) {
        int k0 = a_k0(v, hi, 0);
        int k1 = a_k0(v, hi, 1);
        // A col k: k<24 -> src slice, 24<=k<48 -> dst slice, else 0
        int o0 = (k0 < 24) ? (t * TI + k0) : (96 + t * TI + (k0 - 24));
        A0.u[v] = *(const unsigned*)(rowp + o0);   // k0<48 always in chunk0
        if (k1 < 48) {
          int o1 = (k1 < 24) ? (t * TI + k1) : (96 + t * TI + (k1 - 24));
          A1.u[v] = *(const unsigned*)(rowp + o1);
        } else {
          A1.u[v] = 0u;
        }
      }
      const _Float16* Bt = &BwS[t][0];
      acc0 = wmma16(A0.v, ldb_lds(Bt, 0, 0, BROW, lane), acc0);
      acc0 = wmma16(A1.v, ldb_lds(Bt, 1, 0, BROW, lane), acc0);
      acc1 = wmma16(A0.v, ldb_lds(Bt, 0, 1, BROW, lane), acc1);
      acc1 = wmma16(A1.v, ldb_lds(Bt, 1, 1, BROW, lane), acc1);

      // scatter: 4 atomics per msg element into dst node's aggregates
#pragma unroll
      for (int r = 0; r < 8; ++r) {
        int m = r + 8 * hi;
        int d = __shfl(dReg, m, 32);
        size_t b0 = (size_t)d * DIM + t * TI;
        float v0 = acc0[r];
        atomicAdd(&sum[b0 + lo], v0);
        atomicAdd(&sq[b0 + lo], v0 * v0);
        atomicMax(&mx[b0 + lo], f2k(v0));
        atomicMin(&mn[b0 + lo], f2k(v0));
        if (lo < 8) {
          float v1 = acc1[r];
          size_t c1 = b0 + 16 + lo;
          atomicAdd(&sum[c1], v1);
          atomicAdd(&sq[c1], v1 * v1);
          atomicMax(&mx[c1], f2k(v1));
          atomicMin(&mn[c1], f2k(v1));
        }
      }
    }
  }
}

// ---------------- kernel 4: node update GEMM (U) ----------------
#define NROWS 484   // f32 row stride: 16 rows * 484 * 4B = ~31KB, conflict-free
__global__ __launch_bounds__(32) void node_kernel(
    const float* __restrict__ h, const float* __restrict__ sum, const float* __restrict__ sq,
    const unsigned* __restrict__ mx, const unsigned* __restrict__ mn, const float* __restrict__ deg,
    const float* __restrict__ U_w, const float* __restrict__ U_b, float* __restrict__ u) {
  __shared__ __align__(16) float aggS[16 * NROWS];  // per row m: [ht 96 | mean 96 | max 96 | min 96 | std 96]

  const int lane = threadIdx.x;
  const int hi = lane >> 4, lo = lane & 15;
  const int ntiles = N_NODES / 16;

  for (int tile = blockIdx.x; tile < ntiles; tile += gridDim.x) {
    const int base = tile * 16;

    // stage aggregates for 16 nodes
#pragma unroll 4
    for (int i = 0; i < 48; ++i) {
      int c = lane + 32 * i;              // 1536 = 16*96
      int m = c / 96, col = c % 96;
      int node = base + m;
      size_t gi = (size_t)node * DIM + col;
      float d  = deg[node];
      float ds = fmaxf(d, 1.f);
      float s  = sum[gi], q = sq[gi];
      float mean = s / ds;
      float var  = fmaxf(q / ds - mean * mean, 0.f);
      float sd   = sqrtf(var + 1e-30f);
      float vmx  = k2f(mx[gi]);
      float vmn  = k2f(mn[gi]);
      if (d <= 0.f) { mean = 0.f; sd = 0.f; vmx = 0.f; vmn = 0.f; }
      float* row = aggS + m * NROWS;
      row[col]           = h[gi];
      row[96 + 0 * 96 + col] = mean;
      row[96 + 1 * 96 + col] = vmx;
      row[96 + 2 * 96 + col] = vmn;
      row[96 + 3 * 96 + col] = sd;
    }

    // per-lane degree scalers (A row for this lane is m = lo)
    float dm = deg[base + lo];
    float logd = logf(fmaxf(dm, 1.f) + 1.f);
    float s1 = logd / DELTA, s2 = DELTA / logd;

    const float* row = aggS + lo * NROWS;
#pragma unroll
    for (int t = 0; t < NTOW; ++t) {
      v8f acc0, acc1;
      float b0 = U_b[t * TI + lo];
      float b1 = (lo < 8) ? U_b[t * TI + 16 + lo] : 0.f;
#pragma unroll
      for (int r = 0; r < 8; ++r) { acc0[r] = b0; acc1[r] = b1; }

#pragma unroll
      for (int c = 0; c < 10; ++c) {      // K = 312 padded to 320
        F16x16 A;
#pragma unroll
        for (int v = 0; v < 8; ++v) {
          int k = a_k0(v, hi, c);
          float x0, x1;
          if (k < 24) {                                    // ht block
            x0 = row[t * TI + k]; x1 = row[t * TI + k + 1];
          } else if (k < KIN_U) {                          // agg blocks
            int bb = (k - 24) / 24, j = (k - 24) % 24;
            int a = bb & 3, sI = bb >> 2;
            float scl = (sI == 0) ? 1.f : ((sI == 1) ? s1 : s2);
            const float* p = row + 96 + a * 96 + t * TI + j;
            x0 = p[0] * scl; x1 = p[1] * scl;
          } else { x0 = 0.f; x1 = 0.f; }
          A.h[2 * v]     = (_Float16)x0;
          A.h[2 * v + 1] = (_Float16)x1;
        }
        acc0 = wmma16(A.v, load_bfrag(U_w + (size_t)t * KIN_U * TI, c, 0, KIN_U, TI, TI, 1, lane), acc0);
        acc1 = wmma16(A.v, load_bfrag(U_w + (size_t)t * KIN_U * TI, c, 1, KIN_U, TI, TI, 1, lane), acc1);
      }

#pragma unroll
      for (int r = 0; r < 8; ++r) {
        int node = base + r + 8 * hi;
        u[(size_t)node * DIM + t * TI + lo] = acc0[r];
        if (lo < 8) u[(size_t)node * DIM + t * TI + 16 + lo] = acc1[r];
      }
    }
  }
}

// ---------------- kernel 5: per-column stats -> BN affine (deterministic tree reduce) ----------------
__global__ __launch_bounds__(256) void colstats_kernel(const float* __restrict__ X,
                                                       const float* __restrict__ g, const float* __restrict__ b,
                                                       float* __restrict__ scale, float* __restrict__ shift) {
  __shared__ float sS[256], sQ[256];
  const int c = blockIdx.x;   // 96 columns
  const int tid = threadIdx.x;
  float s = 0.f, q = 0.f;
  for (int r = tid; r < N_NODES; r += 256) {
    float v = X[(size_t)r * DIM + c];
    s += v; q += v * v;
  }
  sS[tid] = s; sQ[tid] = q;
  __syncthreads();
  for (int w = 128; w > 0; w >>= 1) {
    if (tid < w) { sS[tid] += sS[tid + w]; sQ[tid] += sQ[tid + w]; }
    __syncthreads();
  }
  if (tid == 0) {
    float mean = sS[0] / (float)N_NODES;
    float var  = fmaxf(sQ[0] / (float)N_NODES - mean * mean, 0.f);
    float rstd = rsqrtf(var + BN_EPS);
    float sc = g[c] * rstd;
    scale[c] = sc;
    shift[c] = b[c] - mean * sc;
  }
}

// ---------------- kernel 6: BN(u) -> mixing GEMM -> leaky ----------------
#define MROW 98   // f16 row stride for staged mix weights (196B -> conflict-free)
__global__ __launch_bounds__(128) void mix_kernel(const float* __restrict__ u,
                                                  const float* __restrict__ scaleU, const float* __restrict__ shiftU,
                                                  const float* __restrict__ mix_w, const float* __restrict__ mix_b,
                                                  float* __restrict__ hm) {
  __shared__ __align__(16) _Float16 BmS[96 * MROW];   // 18.8 KB: Bm[n][K] = mix_w[n*96 + K]

  const int lane = threadIdx.x & 31;
  const int wave = threadIdx.x >> 5;
  const int hi = lane >> 4, lo = lane & 15;
  const int ntiles = N_NODES / 16;

  // stage mix_w once per block (hc @ mix_w.T -> B[k][o] = mix_w[o*96+k], K-major rows)
  for (int idx = threadIdx.x; idx < 96 * 96; idx += 128) {
    int n = idx / 96, K = idx % 96;
    BmS[n * MROW + K] = (_Float16)mix_w[idx];
  }
  __syncthreads();

  float mb[6];
#pragma unroll
  for (int nt = 0; nt < 6; ++nt) mb[nt] = mix_b[nt * 16 + lo];

  for (int tile = blockIdx.x * 4 + wave; tile < ntiles; tile += gridDim.x * 4) {
    const int base = tile * 16;
    const float* urow = u + (size_t)(base + lo) * DIM;

    v8f acc[6];
#pragma unroll
    for (int nt = 0; nt < 6; ++nt)
#pragma unroll
      for (int r = 0; r < 8; ++r) acc[nt][r] = mb[nt];

#pragma unroll
    for (int c = 0; c < 3; ++c) {        // K = 96
      F16x16 A;
#pragma unroll
      for (int v = 0; v < 8; ++v) {
        int k = a_k0(v, hi, c);          // 0..95, always in range
        float2 xv = *(const float2*)&urow[k];
        float2 sv = *(const float2*)&scaleU[k];
        float2 hv = *(const float2*)&shiftU[k];
        A.h[2 * v]     = (_Float16)(xv.x * sv.x + hv.x);
        A.h[2 * v + 1] = (_Float16)(xv.y * sv.y + hv.y);
      }
#pragma unroll
      for (int nt = 0; nt < 6; ++nt)
        acc[nt] = wmma16(A.v, ldb_lds(BmS, c, nt, MROW, lane), acc[nt]);
    }

#pragma unroll
    for (int nt = 0; nt < 6; ++nt)
#pragma unroll
      for (int r = 0; r < 8; ++r) {
        int node = base + r + 8 * hi;
        hm[(size_t)node * DIM + nt * 16 + lo] = leaky(acc[nt][r]);
      }
  }
}

// ---------------- kernel 7: final BN + leaky + residual ----------------
__global__ __launch_bounds__(256) void final_kernel(const float* __restrict__ h, const float* __restrict__ hm,
                                                    const float* __restrict__ scaleF, const float* __restrict__ shiftF,
                                                    float* __restrict__ out) {
  const size_t n = (size_t)N_NODES * DIM;
  for (size_t i = blockIdx.x * (size_t)blockDim.x + threadIdx.x; i < n;
       i += (size_t)gridDim.x * blockDim.x) {
    int c = (int)(i % DIM);
    out[i] = h[i] + leaky(hm[i] * scaleF[c] + shiftF[c]);
  }
}

// ---------------- host launcher ----------------
extern "C" void kernel_launch(void* const* d_in, const int* in_sizes, int n_in,
                              void* d_out, int out_size, void* d_ws, size_t ws_size,
                              hipStream_t stream) {
  const float* h      = (const float*)d_in[0];
  const int*   src    = (const int*)d_in[1];
  const int*   dst    = (const int*)d_in[2];
  const float* M_w    = (const float*)d_in[3];
  const float* M_b    = (const float*)d_in[4];
  const float* U_w    = (const float*)d_in[5];
  const float* U_b    = (const float*)d_in[6];
  const float* bn_t_g = (const float*)d_in[7];
  const float* bn_t_b = (const float*)d_in[8];
  const float* mix_w  = (const float*)d_in[9];
  const float* mix_b  = (const float*)d_in[10];
  const float* bn_g   = (const float*)d_in[11];
  const float* bn_b   = (const float*)d_in[12];
  float* out = (float*)d_out;

  size_t off = 0;
  auto carve = [&](size_t bytes) -> void* {
    void* p = (char*)d_ws + off;
    off += (bytes + 255) & ~(size_t)255;
    return p;
  };
  const size_t NC = (size_t)N_NODES * DIM;
  float*    sum    = (float*)carve(NC * 4);
  float*    sq     = (float*)carve(NC * 4);
  unsigned* mx     = (unsigned*)carve(NC * 4);
  unsigned* mn     = (unsigned*)carve(NC * 4);
  float*    deg    = (float*)carve((size_t)N_NODES * 4);
  float*    u      = (float*)carve(NC * 4);
  float*    hm     = (float*)carve(NC * 4);
  float*    scaleU = (float*)carve(DIM * 4);
  float*    shiftU = (float*)carve(DIM * 4);
  float*    scaleF = (float*)carve(DIM * 4);
  float*    shiftF = (float*)carve(DIM * 4);

  init_kernel<<<2048, 256, 0, stream>>>(sum, sq, mx, mn, deg);
  deg_kernel<<<2048, 256, 0, stream>>>(dst, deg);
  edge_kernel<<<4096, EWAVES * 32, 0, stream>>>(h, src, dst, M_w, M_b, sum, sq, mx, mn);
  node_kernel<<<3125, 32, 0, stream>>>(h, sum, sq, mx, mn, deg, U_w, U_b, u);
  colstats_kernel<<<96, 256, 0, stream>>>(u, bn_t_g, bn_t_b, scaleU, shiftU);
  mix_kernel<<<1024, 128, 0, stream>>>(u, scaleU, shiftU, mix_w, mix_b, hm);
  colstats_kernel<<<96, 256, 0, stream>>>(hm, bn_g, bn_b, scaleF, shiftF);
  final_kernel<<<2048, 256, 0, stream>>>(h, hm, scaleF, shiftF, out);
}
